// DYNARCLOSS_69295002354045
// MI455X (gfx1250) — matrix operand order repaired and checked by reading, not compiled, not run on hardware
//
#include <hip/hip_runtime.h>
#include <hip/hip_bf16.h>
#include <math.h>

// ---------------------------------------------------------------------------
// DYNARCLOSS for MI455X (gfx1250, wave32, WMMA)
//
// out[b,c] = cos(arccos(logits[b,c]) [+ margin(b) if c==labels[b]]) * 64
//          = logits[b,c]*64 everywhere except one element per row.
// margin(b) needs rowmax_c!=lab( clip(W[lab_b]·W[c], -1, 1) ).
// GEMM via V_WMMA_F32_16X16X32_BF16, 3-term bf16 split (hi/lo) for ~fp32
// accuracy. Each wave owns TWO 16-row M-tiles so every B fragment feeds
// 2x the WMMAs (24 wmma : 16 loads per N-tile) and L2 B-traffic halves.
// GEMM runs BEFORE the 1.6GB streaming scale pass (which is non-temporal)
// so the 51MB of split weights stay resident in the 192MB L2.
// ---------------------------------------------------------------------------

typedef __attribute__((ext_vector_type(16))) __bf16          v16bf;
typedef __attribute__((ext_vector_type(16))) unsigned short  v16u;
typedef __attribute__((ext_vector_type(8)))  unsigned short  u16x8;
typedef __attribute__((ext_vector_type(8)))  float           v8f;
typedef __attribute__((ext_vector_type(4)))  float           v4f;

#define S_SCALE 64.0f
#define K1C 1.0f
#define K2C 0.1f
#define K3C 0.4f

static __device__ __forceinline__ unsigned short bf16_rn(float x) {
  unsigned u = __float_as_uint(x);
  u += 0x7fffu + ((u >> 16) & 1u);
  return (unsigned short)(u >> 16);
}
// order-preserving float -> uint key (atomicMax over floats incl. negatives)
static __device__ __forceinline__ unsigned ordkey(float f) {
  int i = __float_as_int(f);
  return (i < 0) ? ~(unsigned)i : ((unsigned)i | 0x80000000u);
}
static __device__ __forceinline__ float orddec(unsigned u) {
  int i = (u & 0x80000000u) ? (int)(u & 0x7fffffffu) : (int)~u;
  return __int_as_float(i);
}

// --- split fp32 weights into bf16 hi + bf16 lo ------------------------------
__global__ void split_bf16_kernel(const float* __restrict__ w,
                                  unsigned short* __restrict__ hi,
                                  unsigned short* __restrict__ lo,
                                  long long n) {
  long long i = (long long)blockIdx.x * blockDim.x + threadIdx.x;
  long long stride = (long long)gridDim.x * blockDim.x;
  for (; i < n; i += stride) {
    float x = w[i];
    unsigned short h = bf16_rn(x);
    float hf = __uint_as_float((unsigned)h << 16);
    hi[i] = h;
    lo[i] = bf16_rn(x - hf);
  }
}

// --- gather A = W[labels] and split -----------------------------------------
__global__ void gather_split_kernel(const float* __restrict__ w,
                                    const int* __restrict__ labels,
                                    unsigned short* __restrict__ ahi,
                                    unsigned short* __restrict__ alo,
                                    int D, int n) {
  int i = blockIdx.x * blockDim.x + threadIdx.x;
  if (i >= n) return;
  int b = i / D, k = i - b * D;
  int lab = labels[b];
  if (lab < 0) lab = 0;
  float x = w[(size_t)lab * D + k];
  unsigned short h = bf16_rn(x);
  float hf = __uint_as_float((unsigned)h << 16);
  ahi[i] = h;
  alo[i] = bf16_rn(x - hf);
}

__global__ void init_rowmax_kernel(unsigned* __restrict__ rm, int n) {
  int i = blockIdx.x * blockDim.x + threadIdx.x;
  if (i < n) rm[i] = 0x3F7FFFFFu;  // ordkey(-4.0f)
}

// --- bulk: out = logits * S; streaming, non-temporal (don't thrash L2) ------
__global__ void scale_kernel(const v4f* __restrict__ in, v4f* __restrict__ out,
                             long long n4) {
  long long i = (long long)blockIdx.x * blockDim.x + threadIdx.x;
  long long stride = (long long)gridDim.x * blockDim.x;
  for (; i < n4; i += stride) {
    v4f v = __builtin_nontemporal_load(&in[i]);
    v = v * S_SCALE;
    __builtin_nontemporal_store(v, &out[i]);
  }
}
__global__ void scale_tail_kernel(const float* __restrict__ in,
                                  float* __restrict__ out,
                                  long long start, long long n) {
  long long i = start + blockIdx.x * blockDim.x + threadIdx.x;
  if (i < n) out[i] = in[i] * S_SCALE;
}

// --- WMMA GEMM (A=W[labels] hi/lo, B=W^T hi/lo) fused with row-max ----------
static __device__ __forceinline__ v16bf frag16(const unsigned short* p) {
  u16x8 a = *(const u16x8*)(p);
  u16x8 b = *(const u16x8*)(p + 8);
  v16u u = __builtin_shufflevector(a, b, 0, 1, 2, 3, 4, 5, 6, 7,
                                         8, 9, 10, 11, 12, 13, 14, 15);
  return __builtin_bit_cast(v16bf, u);
}
// A fragment pair (hi,lo) for one 16x32 k-chunk, contiguous k = 16*kh + s
static __device__ __forceinline__ v16bf fragA(const unsigned short* row,
                                              int off) {
  u16x8 h0 = *(const u16x8*)(row + off);
  u16x8 h1 = *(const u16x8*)(row + off + 8);
  v16u u = __builtin_shufflevector(h0, h1, 0, 1, 2, 3, 4, 5, 6, 7,
                                           8, 9, 10, 11, 12, 13, 14, 15);
  return __builtin_bit_cast(v16bf, u);
}

__global__ __launch_bounds__(256)
void wmma_rowmax_kernel(const unsigned short* __restrict__ Whi,
                        const unsigned short* __restrict__ Wlo,
                        const unsigned short* __restrict__ Ahi,
                        const unsigned short* __restrict__ Alo,
                        const int* __restrict__ labels,
                        unsigned* __restrict__ rowmax,
                        int D, int ntiles, int tiles_per_chunk) {
  const int lane = threadIdx.x & 31;
  const int wave = threadIdx.x >> 5;
  const int rowbase0 = blockIdx.y * 256 + wave * 32;  // two M-tiles per wave
  const int rowbase1 = rowbase0 + 16;
  const int r  = lane & 15;  // A-row / B-col / D-col within tile
  const int kh = lane >> 4;  // K-half selector

  // A fragments for both M-tiles, all 4 k-chunks, hi+lo: 128 VGPRs resident
  v16bf a0h[4], a0l[4], a1h[4], a1l[4];
  {
    const unsigned short* p0h = Ahi + (size_t)(rowbase0 + r) * D;
    const unsigned short* p0l = Alo + (size_t)(rowbase0 + r) * D;
    const unsigned short* p1h = Ahi + (size_t)(rowbase1 + r) * D;
    const unsigned short* p1l = Alo + (size_t)(rowbase1 + r) * D;
#pragma unroll
    for (int kc = 0; kc < 4; ++kc) {
      const int off = kc * 32 + kh * 16;
      a0h[kc] = fragA(p0h, off);
      a0l[kc] = fragA(p0l, off);
      a1h[kc] = fragA(p1h, off);
      a1l[kc] = fragA(p1l, off);
    }
  }

  int labr0[8], labr1[8];
#pragma unroll
  for (int v = 0; v < 8; ++v) {
    labr0[v] = labels[rowbase0 + kh * 8 + v];
    labr1[v] = labels[rowbase1 + kh * 8 + v];
  }

  float mx0[8], mx1[8];
#pragma unroll
  for (int v = 0; v < 8; ++v) { mx0[v] = -3.0e38f; mx1[v] = -3.0e38f; }

  int n0 = blockIdx.x * tiles_per_chunk;
  int n1 = n0 + tiles_per_chunk;
  if (n1 > ntiles) n1 = ntiles;

  for (int nt = n0; nt < n1; ++nt) {
    const int col = nt * 16 + r;
    // B = W^T: column 'col' of B is row 'col' of W -> K contiguous in memory
    const unsigned short* pbh = Whi + (size_t)col * D + kh * 16;
    const unsigned short* pbl = Wlo + (size_t)col * D + kh * 16;
    v8f acc0 = {0.f, 0.f, 0.f, 0.f, 0.f, 0.f, 0.f, 0.f};
    v8f acc1 = {0.f, 0.f, 0.f, 0.f, 0.f, 0.f, 0.f, 0.f};
#pragma unroll
    for (int kc = 0; kc < 4; ++kc) {
      const int off = kc * 32;
      v16bf bhi = frag16(pbh + off);
      v16bf blo = frag16(pbl + off);
      // 3-term split per M-tile, interleaved: two independent WMMA chains
      acc0 = __builtin_amdgcn_wmma_f32_16x16x32_bf16(false, a0h[kc], false, bhi,
                                                     (short)0, acc0, false, false);
      acc1 = __builtin_amdgcn_wmma_f32_16x16x32_bf16(false, a1h[kc], false, bhi,
                                                     (short)0, acc1, false, false);
      acc0 = __builtin_amdgcn_wmma_f32_16x16x32_bf16(false, a0h[kc], false, blo,
                                                     (short)0, acc0, false, false);
      acc1 = __builtin_amdgcn_wmma_f32_16x16x32_bf16(false, a1h[kc], false, blo,
                                                     (short)0, acc1, false, false);
      acc0 = __builtin_amdgcn_wmma_f32_16x16x32_bf16(false, a0l[kc], false, bhi,
                                                     (short)0, acc0, false, false);
      acc1 = __builtin_amdgcn_wmma_f32_16x16x32_bf16(false, a1l[kc], false, bhi,
                                                     (short)0, acc1, false, false);
    }
    // D layout: lane owns column 'col'; VGPR v -> row kh*8+v. Skip true class.
#pragma unroll
    for (int v = 0; v < 8; ++v) {
      if (labr0[v] != col) mx0[v] = fmaxf(mx0[v], acc0[v]);
      if (labr1[v] != col) mx1[v] = fmaxf(mx1[v], acc1[v]);
    }
    __syncthreads();  // keep 8 waves walking identical B tiles (WGP$ reuse)
  }

  // reduce across the 16 lanes sharing each row set
#pragma unroll
  for (int v = 0; v < 8; ++v) {
#pragma unroll
    for (int m = 1; m < 16; m <<= 1) {
      mx0[v] = fmaxf(mx0[v], __shfl_xor(mx0[v], m, 32));
      mx1[v] = fmaxf(mx1[v], __shfl_xor(mx1[v], m, 32));
    }
  }
  if (r == 0) {
#pragma unroll
    for (int v = 0; v < 8; ++v) {
      atomicMax(&rowmax[rowbase0 + kh * 8 + v], ordkey(mx0[v]));
      atomicMax(&rowmax[rowbase1 + kh * 8 + v], ordkey(mx1[v]));
    }
  }
}

// --- per-row margin + single-element fixup ----------------------------------
__global__ void finalize_kernel(const float* __restrict__ logits,
                                const int* __restrict__ labels,
                                const unsigned* __restrict__ rowmax,
                                float* __restrict__ out, int B, int C) {
  int b = blockIdx.x * blockDim.x + threadIdx.x;
  if (b >= B) return;
  int lab = labels[b];
  if (lab < 0) return;  // invalid: bulk pass already wrote logits*S
  float m = orddec(rowmax[b]);
  m = fminf(1.0f, fmaxf(-1.0f, m));
  float theta_m = acosf(m);
  float d = theta_m - K1C;
  float smooth = 0.03f * K3C / (1.0f + powf(fabsf(d) * 20.0f, 1.1f));
  float margin = fmaxf(d, 0.0f) * K2C + K3C + smooth;
  float t = acosf(logits[(size_t)b * C + lab]);
  out[(size_t)b * C + lab] = cosf(t + margin) * S_SCALE;
}

// ---------------------------------------------------------------------------
extern "C" void kernel_launch(void* const* d_in, const int* in_sizes, int n_in,
                              void* d_out, int out_size, void* d_ws,
                              size_t ws_size, hipStream_t stream) {
  const float* logits = (const float*)d_in[0];
  const int*   labels = (const int*)d_in[1];
  const float* wnorm  = (const float*)d_in[2];

  const int B = in_sizes[1];
  const long long BC = (long long)in_sizes[0];
  const int C = (int)(BC / B);
  const int D = (int)(in_sizes[2] / C);  // 128

  // workspace layout (~52.3 MB)
  char* p = (char*)d_ws;
  auto take = [&](size_t bytes) {
    void* q = (void*)p;
    p += (bytes + 255) & ~(size_t)255;
    return q;
  };
  const size_t CD = (size_t)C * D, BD = (size_t)B * D;
  unsigned short* Whi = (unsigned short*)take(CD * 2);
  unsigned short* Wlo = (unsigned short*)take(CD * 2);
  unsigned short* Ahi = (unsigned short*)take(BD * 2);
  unsigned short* Alo = (unsigned short*)take(BD * 2);
  unsigned* rowmax = (unsigned*)take((size_t)B * 4);

  split_bf16_kernel<<<4096, 256, 0, stream>>>(wnorm, Whi, Wlo, (long long)CD);
  gather_split_kernel<<<((int)BD + 255) / 256, 256, 0, stream>>>(
      wnorm, labels, Ahi, Alo, D, (int)BD);
  init_rowmax_kernel<<<(B + 255) / 256, 256, 0, stream>>>(rowmax, B);

  // GEMM first: split weights are L2-resident, untouched by streaming pass
  const int ntiles = C / 16;            // 6250
  const int tiles_per_chunk = 25;
  const int chunks = (ntiles + tiles_per_chunk - 1) / tiles_per_chunk;  // 250
  dim3 grid(chunks, B / 256);           // (250, 8) blocks; 8 waves x 32 rows
  wmma_rowmax_kernel<<<grid, 256, 0, stream>>>(Whi, Wlo, Ahi, Alo, labels,
                                               rowmax, D, ntiles,
                                               tiles_per_chunk);

  const long long n4 = BC / 4;
  scale_kernel<<<8192, 256, 0, stream>>>((const v4f*)logits, (v4f*)d_out, n4);
  if (BC & 3) {
    scale_tail_kernel<<<1, 64, 0, stream>>>((const float*)logits, (float*)d_out,
                                            n4 * 4, BC);
  }

  finalize_kernel<<<(B + 255) / 256, 256, 0, stream>>>(logits, labels, rowmax,
                                                       (float*)d_out, B, C);
}
